// GRNNCell_6133213299289
// MI455X (gfx1250) — matrix operand order repaired and checked by reading, not compiled, not run on hardware
//
#include <hip/hip_runtime.h>

typedef __bf16 bf16;
typedef __attribute__((ext_vector_type(16))) __bf16 v16bf;
typedef __attribute__((ext_vector_type(8)))  __bf16 v8bf;
typedef __attribute__((ext_vector_type(8)))  float  v8f;

#define Gs   12
#define Hs   512
#define Bsz  512
#define GH   6144      // Gs*Hs
#define BK   32
#define BM   128
#define BN   128
#define LDST 40        // padded LDS row stride in bf16 elems (80 bytes)

// ---------------- f32 -> bf16 elementwise convert ----------------
__global__ void cvt_act(const float* __restrict__ src, bf16* __restrict__ dst, int n) {
    int i = (blockIdx.x * blockDim.x + threadIdx.x) * 4;
    if (i < n) {
        float4 v = *(const float4*)(src + i);
        dst[i + 0] = (bf16)v.x;
        dst[i + 1] = (bf16)v.y;
        dst[i + 2] = (bf16)v.z;
        dst[i + 3] = (bf16)v.w;
    }
}

// ------------- W (g,i,e) f32 -> WT (g,e,i) bf16 (tiled transpose) -------------
__global__ void cvt_wt(const float* __restrict__ W, bf16* __restrict__ WT) {
    __shared__ float t[32][33];
    const int g  = blockIdx.z;
    const int e0 = blockIdx.x * 32;
    const int i0 = blockIdx.y * 32;
    const float* Wg  = W  + (size_t)g * Hs * Hs;
    bf16*        WTg = WT + (size_t)g * Hs * Hs;
    const int tx = threadIdx.x, ty = threadIdx.y;
#pragma unroll
    for (int j = 0; j < 32; j += 8)
        t[ty + j][tx] = Wg[(size_t)(i0 + ty + j) * Hs + (e0 + tx)];
    __syncthreads();
#pragma unroll
    for (int j = 0; j < 32; j += 8)
        WTg[(size_t)(e0 + ty + j) * Hs + (i0 + tx)] = (bf16)t[tx][ty + j];
}

// ---------------- main block-circulant GEMM + bias + tanh ----------------
__global__ __launch_bounds__(256) void grnn_gemm(
    const bf16* __restrict__ Ain,   // (B, GH)
    const bf16* __restrict__ Ahid,  // (B, GH)
    const bf16* __restrict__ WwT,   // (G, Hs e, Hs i)
    const bf16* __restrict__ WhT,   // (G, Hs e, Hs i)
    const float* __restrict__ bw,
    const float* __restrict__ bh,
    float* __restrict__ out)        // (B, G, Hs)
{
    __shared__ __align__(16) bf16 As[2][BM * LDST];
    __shared__ __align__(16) bf16 Bs[2][BN * LDST];

    const int tid = threadIdx.x;
    const int h   = blockIdx.z;
    const int m0  = blockIdx.y * BM;
    const int n0  = blockIdx.x * BN;

    const int wid    = tid >> 5;
    const int lane   = tid & 31;
    const int lr     = lane & 15;
    const int laneHi = lane >> 4;          // 0 or 1

    const int wm = wid >> 1;               // 0..3 -> 32-row strip (M)
    const int wn = wid & 1;                // 0..1 -> 64-col strip (N)

    // staging: 4 threads cover one row's 32 K-elems (4 x v8bf); rows r and r+64
    const int srow = tid >> 2;             // 0..63
    const int skb  = (tid & 3) * 8;        // 0,8,16,24 (bf16 elems)

    // compile-time-constant LDS offsets (element units)
    const int aSt0 = srow * LDST + skb;
    const int aSt1 = (srow + 64) * LDST + skb;
    const int aRd  = (wm * 32 + lr) * LDST + laneHi * 8;
    const int bRd  = (wn * 64 + lr) * LDST + laneHi * 8;

    v8f acc[2][4];
#pragma unroll
    for (int i = 0; i < 2; ++i)
#pragma unroll
        for (int j = 0; j < 4; ++j)
#pragma unroll
            for (int r = 0; r < 8; ++r) acc[i][j][r] = 0.0f;

    v8bf ra0, ra1, rb0, rb1;
    const bf16* Aseg;   // per-segment A base for row srow (chunk skb)
    const bf16* Bseg;   // per-segment B base for row srow (chunk skb)

    auto loadStep = [&](int j) {
        const int o = j * BK;
        ra0 = *(const v8bf*)(Aseg + o);
        ra1 = *(const v8bf*)(Aseg + (size_t)64 * GH + o);
        rb0 = *(const v8bf*)(Bseg + o);
        rb1 = *(const v8bf*)(Bseg + (size_t)64 * Hs + o);
    };

    union Frag { v16bf v; v8bf h2[2]; };

    auto computeBuf = [&](int buf) {
        Frag a[2], b[4];
        const bf16* Ab = &As[buf][0];
        const bf16* Bb = &Bs[buf][0];
#pragma unroll
        for (int ns = 0; ns < 4; ++ns) {
            // B 32x16 layout: lo lanes K0-7 & K16-23, hi lanes K8-15 & K24-31
            b[ns].h2[0] = *(const v8bf*)(Bb + bRd + ns * 16 * LDST);
            b[ns].h2[1] = *(const v8bf*)(Bb + bRd + ns * 16 * LDST + 16);
        }
#pragma unroll
        for (int ms = 0; ms < 2; ++ms) {
            // A 16x32 layout: lo lanes K0-7 & K16-23, hi lanes K8-15 & K24-31
            a[ms].h2[0] = *(const v8bf*)(Ab + aRd + ms * 16 * LDST);
            a[ms].h2[1] = *(const v8bf*)(Ab + aRd + ms * 16 * LDST + 16);
        }
#pragma unroll
        for (int ms = 0; ms < 2; ++ms)
#pragma unroll
            for (int ns = 0; ns < 4; ++ns)
                acc[ms][ns] = __builtin_amdgcn_wmma_f32_16x16x32_bf16(
                    false, a[ms].v, false, b[ns].v,
                    (short)0, acc[ms][ns], false, false);
    };

#define STORE_BUF(buf)                                          \
    do {                                                        \
        *(v8bf*)&As[buf][aSt0] = ra0;                           \
        *(v8bf*)&As[buf][aSt1] = ra1;                           \
        *(v8bf*)&Bs[buf][aSt0] = rb0;                           \
        *(v8bf*)&Bs[buf][aSt1] = rb1;                           \
    } while (0)

    // 24 segments: (s = input/hidden) x (g' = 0..11); 16 K-steps each.
#pragma unroll 1
    for (int seg = 0; seg < 2 * Gs; ++seg) {
        const int s  = seg >= Gs;
        const int gp = seg - s * Gs;
        int gsrc = h - gp; if (gsrc < 0) gsrc += Gs;
        Aseg = (s ? Ahid : Ain) + (size_t)(m0 + srow) * GH + gp * Hs + skb;
        Bseg = (s ? WhT : WwT) + ((size_t)gsrc * Hs + (n0 + srow)) * (size_t)Hs + skb;

        loadStep(0);
        STORE_BUF(0);
        __syncthreads();

#pragma unroll 1
        for (int j = 0; j < 14; j += 2) {
            loadStep(j + 1);
            computeBuf(0);
            STORE_BUF(1);
            __syncthreads();
            loadStep(j + 2);
            computeBuf(1);
            STORE_BUF(0);
            __syncthreads();
        }
        // tail: j = 14, 15
        loadStep(15);
        computeBuf(0);
        STORE_BUF(1);
        __syncthreads();
        computeBuf(1);
        __syncthreads();
    }
#undef STORE_BUF

    // Epilogue: C/D layout -> lane = N, VGPR r = M (+8 for hi lanes)
#pragma unroll
    for (int ms = 0; ms < 2; ++ms) {
#pragma unroll
        for (int ns = 0; ns < 4; ++ns) {
            const int e = n0 + wn * 64 + ns * 16 + lr;
            const float bias = bw[e] + bh[e];
#pragma unroll
            for (int r = 0; r < 8; ++r) {
                const int m = m0 + wm * 32 + ms * 16 + laneHi * 8 + r;
                out[((size_t)m * Gs + h) * Hs + e] = tanhf(acc[ms][ns][r] + bias);
            }
        }
    }
}

extern "C" void kernel_launch(void* const* d_in, const int* in_sizes, int n_in,
                              void* d_out, int out_size, void* d_ws, size_t ws_size,
                              hipStream_t stream) {
    const float* input  = (const float*)d_in[0];
    const float* hidden = (const float*)d_in[1];
    const float* W_w    = (const float*)d_in[2];
    const float* b_w    = (const float*)d_in[3];
    const float* W_h    = (const float*)d_in[4];
    const float* b_h    = (const float*)d_in[5];
    // d_in[6] (perm) is recomputed analytically: gsrc = (h - g') mod 12
    float* out = (float*)d_out;

    bf16* inBF  = (bf16*)d_ws;
    bf16* hidBF = inBF  + (size_t)Bsz * GH;
    bf16* WwT   = hidBF + (size_t)Bsz * GH;
    bf16* WhT   = WwT   + (size_t)Gs * Hs * Hs;
    // total workspace use: 4 * 3,145,728 * 2B = 24 MiB

    const int nAct = Bsz * GH;             // 3,145,728
    cvt_act<<<dim3(nAct / 4 / 256), 256, 0, stream>>>(input,  inBF,  nAct);
    cvt_act<<<dim3(nAct / 4 / 256), 256, 0, stream>>>(hidden, hidBF, nAct);
    cvt_wt<<<dim3(16, 16, Gs), dim3(32, 8), 0, stream>>>(W_w, WwT);
    cvt_wt<<<dim3(16, 16, Gs), dim3(32, 8), 0, stream>>>(W_h, WhT);

    grnn_gemm<<<dim3(Hs / BN, Bsz / BM, Gs), 256, 0, stream>>>(
        inBF, hidBF, WwT, WhT, b_w, b_h, out);
}
